// SlidingWindowAttentionGQA_89867895702177
// MI455X (gfx1250) — matrix operand order repaired and checked by reading, not compiled, not run on hardware
//
#include <hip/hip_runtime.h>

namespace cdna5_swa {

constexpr int S    = 2048;
constexpr int EMB  = 1024;
constexpr int HQN  = 16;
constexpr int HKVN = 4;
constexpr int HD   = 64;
constexpr int WINR = 128;

typedef __attribute__((ext_vector_type(16))) __bf16 v16bf;
typedef __attribute__((ext_vector_type(8)))  float  v8f;

union Frag16 {
  v16bf v;
  uint4 q[2];
  unsigned short u[16];
};

__device__ __forceinline__ unsigned short f2bf(float f) {
  unsigned int u = __float_as_uint(f);
  u += 0x7FFFu + ((u >> 16) & 1u);   // round-to-nearest-even
  return (unsigned short)(u >> 16);
}

// A-matrix fragment (16x32 bf16, MxK): lane half h holds K contiguous
// {h*16 .. h*16+15}; row = lane%16. Two b128 loads.
__device__ __forceinline__ v16bf load_a_frag(const unsigned short* base) {
  const int h8 = ((threadIdx.x >> 4) & 1) << 3;
  Frag16 f;
  f.q[0] = *(const uint4*)(base + 2 * h8);
  f.q[1] = *(const uint4*)(base + 2 * h8 + 8);
  return f.v;
}

// B-matrix fragment (32x16 bf16, KxN): lane half h holds K contiguous
// {h*16 .. h*16+15}; col = lane%16. Two b128 loads (one 32B region).
__device__ __forceinline__ v16bf load_b_frag(const unsigned short* base) {
  const int h16 = ((threadIdx.x >> 4) & 1) << 4;
  Frag16 f;
  f.q[0] = *(const uint4*)(base + h16);
  f.q[1] = *(const uint4*)(base + h16 + 8);
  return f.v;
}

__device__ __forceinline__ v8f wmma_bf16(v16bf a, v16bf b, v8f c) {
  return __builtin_amdgcn_wmma_f32_16x16x32_bf16(
      /*neg_a=*/false, a, /*neg_b=*/false, b,
      /*c_mod=*/(short)0, c, /*reuse_a=*/false, /*reuse_b=*/false);
}

__global__ void cvt_bf16(const float* __restrict__ in,
                         unsigned short* __restrict__ out, int n) {
  int i = blockIdx.x * blockDim.x + threadIdx.x;
  if (i < n) out[i] = f2bf(in[i]);
}

// C = A(MxK) x W(NxK)^T, bf16 inputs, f32 accumulate. One wave -> 32x64 tile
// (2 A-fragments x 4 B-fragments = 8 independent WMMAs per k-step; halves
// weight-fragment traffic vs a 16-row tile and widens the WMMA ILP window).
// mode 0: out bf16 [M][N];  mode 1: out bf16 transposed [N][M];  mode 2: out f32 [M][N]
__global__ void gemm_bf16(const unsigned short* __restrict__ A,
                          const unsigned short* __restrict__ W,
                          void* __restrict__ out,
                          int M, int N, int K, int mode) {
  const int m0   = blockIdx.x << 5;
  const int n0   = blockIdx.y << 6;
  const int lane = threadIdx.x & 31;
  const int col  = lane & 15;
  const int half = lane >> 4;

  const unsigned short* arow0 = A + (size_t)(m0 + col) * K;
  const unsigned short* arow1 = A + (size_t)(m0 + 16 + col) * K;
  const unsigned short* wrow  = W + (size_t)(n0 + col) * K;

  v8f acc[2][4] = {};
  for (int k0 = 0; k0 < K; k0 += 32) {
    v16bf a0 = load_a_frag(arow0 + k0);
    v16bf a1 = load_a_frag(arow1 + k0);
#pragma unroll
    for (int t = 0; t < 4; ++t) {
      v16bf bf = load_b_frag(wrow + (size_t)(t << 4) * K + k0);
      acc[0][t] = wmma_bf16(a0, bf, acc[0][t]);
      acc[1][t] = wmma_bf16(a1, bf, acc[1][t]);
    }
  }

#pragma unroll
  for (int r = 0; r < 2; ++r) {
    const int mr = m0 + (r << 4);
    if (mode == 1) {
      unsigned short* ot = (unsigned short*)out;
#pragma unroll
      for (int t = 0; t < 4; ++t) {
        union { uint4 q; unsigned short u[8]; } p;
#pragma unroll
        for (int v = 0; v < 8; ++v) p.u[v] = f2bf(acc[r][t][v]);
        const int n = n0 + (t << 4) + col;
        *(uint4*)(ot + (size_t)n * M + mr + (half << 3)) = p.q;  // 8 bf16 packed
      }
    } else if (mode == 0) {
      unsigned short* o = (unsigned short*)out;
#pragma unroll
      for (int t = 0; t < 4; ++t)
#pragma unroll
        for (int v = 0; v < 8; ++v)
          o[(size_t)(mr + (half << 3) + v) * N + n0 + (t << 4) + col] = f2bf(acc[r][t][v]);
    } else {
      float* o = (float*)out;
#pragma unroll
      for (int t = 0; t < 4; ++t)
#pragma unroll
        for (int v = 0; v < 8; ++v)
          o[(size_t)(mr + (half << 3) + v) * N + n0 + (t << 4) + col] = acc[r][t][v];
    }
  }
}

// Flash-style sliding-window attention, transposed score layout.
// One wave per (16-query tile, q-head). S^T = K x Q^T so each lane owns one
// query column -> cheap per-query softmax reductions. O^T = V^T x P^T.
__global__ void swa_attn(const unsigned short* __restrict__ Qb,  // [S][EMB]
                         const unsigned short* __restrict__ Kb,  // [S][HKV*HD]
                         const unsigned short* __restrict__ Vt,  // [HKV*HD][S]
                         unsigned short* __restrict__ ctx) {     // [S][EMB]
  const int s0   = blockIdx.x << 4;
  const int hq   = blockIdx.y;
  const int h    = hq >> 2;          // G = 4 query heads per kv head
  const int lane = threadIdx.x & 31;
  const int col  = lane & 15;        // query index within tile
  const int half = lane >> 4;
  const int h8   = half << 3;

  const unsigned short* qrow = Qb + (size_t)(s0 + col) * EMB + hq * HD;
  v16bf bq0 = load_b_frag(qrow);        // d = 0..31
  v16bf bq1 = load_b_frag(qrow + 32);   // d = 32..63

  float m = -3.0e4f;
  float l = 0.f;
  v8f o[4] = {};
  const int sq = s0 + col;

  for (int kb = 0; kb < 9; ++kb) {      // 9 x 32 keys covers [s0-128, s0+159]
    const int j0 = s0 - WINR + (kb << 5);

    v8f st[2];
#pragma unroll
    for (int sub = 0; sub < 2; ++sub) {
      int jr = j0 + (sub << 4) + col;                    // key row (A-matrix row)
      jr = jr < 0 ? 0 : (jr > S - 1 ? S - 1 : jr);      // clamp; masked later
      const unsigned short* krow = Kb + (size_t)jr * (HKVN * HD) + h * HD;
      v8f a = {};
      a = wmma_bf16(load_a_frag(krow),      bq0, a);
      a = wmma_bf16(load_a_frag(krow + 32), bq1, a);
      st[sub] = a;
    }

    // scale + window mask (finite -3e4 so boundary blocks can't NaN)
    float sc[2][8];
    float mb = -3.0e4f;
#pragma unroll
    for (int sub = 0; sub < 2; ++sub)
#pragma unroll
      for (int v = 0; v < 8; ++v) {
        const int j = j0 + (sub << 4) + h8 + v;
        float x = st[sub][v] * 0.125f;   // 1/sqrt(64)
        const bool ok = (j >= 0) & (j < S) & (j >= sq - WINR) & (j <= sq + WINR);
        x = ok ? x : -3.0e4f;
        sc[sub][v] = x;
        mb = fmaxf(mb, x);
      }
    mb = fmaxf(mb, __shfl_xor(mb, 16, 32));
    const float mn = fmaxf(m, mb);
    const float alpha = __expf(m - mn);
    m = mn;

    float e0[8], e1[8];
    float rs = 0.f;
#pragma unroll
    for (int v = 0; v < 8; ++v) {
      e0[v] = __expf(sc[0][v] - mn);
      e1[v] = __expf(sc[1][v] - mn);
      rs += e0[v] + e1[v];
    }
    rs += __shfl_xor(rs, 16, 32);
    l = l * alpha + rs;
#pragma unroll
    for (int t = 0; t < 4; ++t) o[t] = o[t] * alpha;

    // Build P^T B-fragment: B wants K (=j) contiguous {half*16..+15};
    // S^T frags hold j = j0 + sub*16 + half*8 + v -> 8 values from partner half.
    Frag16 pf;
#pragma unroll
    for (int v = 0; v < 8; ++v) {
      const float send = half ? e0[v] : e1[v];
      const float recv = __shfl_xor(send, 16, 32);
      pf.u[v]     = f2bf(half ? recv : e0[v]);
      pf.u[v + 8] = f2bf(half ? e1[v] : recv);
    }

    int c0 = j0 + (half << 4);
    int c1 = j0 + (half << 4) + 8;
    c0 = c0 < 0 ? 0 : (c0 > S - 8 ? S - 8 : c0);  // clamped loads; P==0 there
    c1 = c1 < 0 ? 0 : (c1 > S - 8 ? S - 8 : c1);
#pragma unroll
    for (int t = 0; t < 4; ++t) {
      const unsigned short* vrow = Vt + (size_t)(h * HD + (t << 4) + col) * S;
      Frag16 va;                         // V^T A-fragment, K = j
      va.q[0] = *(const uint4*)(vrow + c0);
      va.q[1] = *(const uint4*)(vrow + c1);
      o[t] = wmma_bf16(va.v, pf.v, o[t]);
    }
  }

  const float inv = 1.0f / l;
  unsigned short* crow = ctx + (size_t)(s0 + col) * EMB + hq * HD;
#pragma unroll
  for (int t = 0; t < 4; ++t) {
    union { uint4 q; unsigned short u[8]; } p;
#pragma unroll
    for (int v = 0; v < 8; ++v) p.u[v] = f2bf(o[t][v] * inv);
    *(uint4*)(crow + (t << 4) + h8) = p.q;  // 8 contiguous d per lane
  }
}

} // namespace cdna5_swa

extern "C" void kernel_launch(void* const* d_in, const int* in_sizes, int n_in,
                              void* d_out, int out_size, void* d_ws, size_t ws_size,
                              hipStream_t stream) {
  using namespace cdna5_swa;
  (void)in_sizes; (void)n_in; (void)out_size; (void)ws_size;

  const float* x  = (const float*)d_in[0];
  const float* Wq = (const float*)d_in[1];
  const float* Wk = (const float*)d_in[2];
  const float* Wv = (const float*)d_in[3];
  const float* Wo = (const float*)d_in[4];

  unsigned short* w   = (unsigned short*)d_ws;
  unsigned short* xb  = w;                                  // [S][EMB]
  unsigned short* wqb = xb  + (size_t)S * EMB;              // [EMB][EMB]
  unsigned short* wkb = wqb + (size_t)EMB * EMB;            // [256][EMB]
  unsigned short* wvb = wkb + (size_t)HKVN * HD * EMB;      // [256][EMB]
  unsigned short* wob = wvb + (size_t)HKVN * HD * EMB;      // [EMB][EMB]
  unsigned short* qb  = wob + (size_t)EMB * EMB;            // [S][EMB]
  unsigned short* kb  = qb  + (size_t)S * EMB;              // [S][256]
  unsigned short* vt  = kb  + (size_t)S * HKVN * HD;        // [256][S] transposed
  unsigned short* ctx = vt  + (size_t)S * HKVN * HD;        // [S][EMB]

  auto cvt = [&](const float* in, unsigned short* out, int n) {
    cvt_bf16<<<(n + 255) / 256, 256, 0, stream>>>(in, out, n);
  };
  cvt(x,  xb,  S * EMB);
  cvt(Wq, wqb, EMB * EMB);
  cvt(Wk, wkb, HKVN * HD * EMB);
  cvt(Wv, wvb, HKVN * HD * EMB);
  cvt(Wo, wob, EMB * EMB);

  // Projections (32-row M-tiles per wave)
  gemm_bf16<<<dim3(S / 32, EMB / 64), 32, 0, stream>>>(xb, wqb, qb, S, EMB, EMB, 0);
  gemm_bf16<<<dim3(S / 32, (HKVN * HD) / 64), 32, 0, stream>>>(xb, wkb, kb, S, HKVN * HD, EMB, 0);
  gemm_bf16<<<dim3(S / 32, (HKVN * HD) / 64), 32, 0, stream>>>(xb, wvb, vt, S, HKVN * HD, EMB, 1);

  // Sliding-window attention (one wave per 16-query tile per q-head)
  swa_attn<<<dim3(S / 16, HQN), 32, 0, stream>>>(qb, kb, vt, ctx);

  // Output projection -> f32 d_out
  gemm_bf16<<<dim3(S / 32, EMB / 64), 32, 0, stream>>>(ctx, wob, (float*)d_out, S, EMB, EMB, 2);
}
